// EnhancedMultiHeadAttention_51161650430488
// MI455X (gfx1250) — matrix operand order repaired and checked by reading, not compiled, not run on hardware
//
#include <hip/hip_runtime.h>
#include <hip/hip_bf16.h>

// ---------------------------------------------------------------------------
// EnhancedMultiHeadAttention for MI455X (gfx1250), wave32 + WMMA bf16.
// B=2, S=2048, E=1024, H=16, D=64.
// All GEMMs on v_wmma_f32_16x16x32_bf16 (f32 accumulate), register-blocked
// 32x64 per wave so the load:WMMA issue ratio is ~1.5:1 instead of 4:1.
// ---------------------------------------------------------------------------

#define BATCH 2
#define SEQ   2048
#define EMBED 1024
#define HEADS 16
#define HDIM  64
#define MROWS (BATCH * SEQ)      // 4096

typedef __attribute__((ext_vector_type(16))) __bf16 bf16x16;
typedef __attribute__((ext_vector_type(8)))  __bf16 bf16x8;
typedef __attribute__((ext_vector_type(8)))  float  f32x8;

__device__ __forceinline__ bf16x16 load_frag(const __bf16* p0, const __bf16* p1) {
    bf16x8 lo = *(const bf16x8*)p0;     // global_load_b128
    bf16x8 hi = *(const bf16x8*)p1;     // global_load_b128
    return __builtin_shufflevector(lo, hi, 0,1,2,3,4,5,6,7,8,9,10,11,12,13,14,15);
}

__device__ __forceinline__ f32x8 wmma_bf16(bf16x16 a, bf16x16 b, f32x8 c) {
    // D = A(16x32 bf16) * B(32x16 bf16) + C(16x16 f32)
    return __builtin_amdgcn_wmma_f32_16x16x32_bf16(
        false, a, false, b, (short)0, c, false, false);
}

// ---------------------------------------------------------------------------
// fp32 -> bf16 elementwise
// ---------------------------------------------------------------------------
__global__ __launch_bounds__(256) void k_f32_to_bf16(const float* __restrict__ in,
                                                     __bf16* __restrict__ out, int n) {
    for (int i = blockIdx.x * blockDim.x + threadIdx.x; i < n; i += gridDim.x * blockDim.x)
        out[i] = (__bf16)in[i];
}

// ---------------------------------------------------------------------------
// W[1024x1024] fp32 row-major -> Wt[n][k] = W[k][n] in bf16 (LDS tiled)
// ---------------------------------------------------------------------------
__global__ __launch_bounds__(256) void k_transpose_bf16(const float* __restrict__ W,
                                                        __bf16* __restrict__ Wt) {
    __shared__ float tile[32][33];
    int bx = blockIdx.x, by = blockIdx.y;
    int tx = threadIdx.x, ty = threadIdx.y;           // (32, 8)
    #pragma unroll
    for (int j = 0; j < 32; j += 8)
        tile[ty + j][tx] = W[(size_t)(by * 32 + ty + j) * EMBED + bx * 32 + tx];
    __syncthreads();
    #pragma unroll
    for (int j = 0; j < 32; j += 8)
        Wt[(size_t)(bx * 32 + ty + j) * EMBED + by * 32 + tx] = (__bf16)tile[tx][ty + j];
}

// ---------------------------------------------------------------------------
// Projection GEMM: C[4096,1024] = A_bf16[M,1024] @ Wt[N][K=1024] + bias.
// Register-blocked: each wave computes a 32(M) x 64(N) macro-tile
// (2 m-subtiles x 4 n-subtiles = 8 WMMAs per 32-wide K step).
// MODE 0: bf16 row-major out.  MODE 1: bf16 V^T out [b][h][d][s].  MODE 2: f32 out.
// ---------------------------------------------------------------------------
template <int MODE>
__global__ __launch_bounds__(256) void k_proj_gemm(const __bf16* __restrict__ A,
                                                   const __bf16* __restrict__ Wt,
                                                   const float* __restrict__ bias,
                                                   void* __restrict__ out) {
    int w    = (blockIdx.x * blockDim.x + threadIdx.x) >> 5;    // 0..2047
    int lane = threadIdx.x & 31;
    int lcol = lane & 15;
    int lhi  = lane >> 4;
    int ng   = w & 15;          // 16 groups of 64 columns
    int mt   = w >> 4;          // 128 groups of 32 rows

    const __bf16* arow0 = A + (size_t)(mt * 32 + lcol) * EMBED;
    const __bf16* arow1 = arow0 + (size_t)16 * EMBED;
    const __bf16* brow[4];
    #pragma unroll
    for (int j = 0; j < 4; ++j)
        brow[j] = Wt + (size_t)(ng * 64 + j * 16 + lcol) * EMBED;

    f32x8 c[2][4] = {};
    for (int kk = 0; kk < EMBED; kk += 32) {
        bf16x16 a0 = load_frag(arow0 + kk + lhi * 16, arow0 + kk + lhi * 16 + 8);
        bf16x16 a1 = load_frag(arow1 + kk + lhi * 16, arow1 + kk + lhi * 16 + 8);
        #pragma unroll
        for (int j = 0; j < 4; ++j) {
            bf16x16 bf = load_frag(brow[j] + kk + lhi * 16, brow[j] + kk + lhi * 16 + 8);
            c[0][j] = wmma_bf16(a0, bf, c[0][j]);
            c[1][j] = wmma_bf16(a1, bf, c[1][j]);
        }
    }

    #pragma unroll
    for (int j = 0; j < 4; ++j) {
        int n = ng * 64 + j * 16 + lcol;
        float bi = bias[n];
        #pragma unroll
        for (int mi = 0; mi < 2; ++mi) {
            #pragma unroll
            for (int r = 0; r < 8; ++r) {
                int m = mt * 32 + mi * 16 + r + lhi * 8;
                float v = c[mi][j][r] + bi;
                if constexpr (MODE == 0) {
                    ((__bf16*)out)[(size_t)m * EMBED + n] = (__bf16)v;
                } else if constexpr (MODE == 1) {
                    int b_ = m >> 11, s = m & 2047, h = n >> 6, d = n & 63;
                    ((__bf16*)out)[((size_t)((b_ * HEADS + h) * HDIM + d)) * SEQ + s] = (__bf16)v;
                } else {
                    ((float*)out)[(size_t)m * EMBED + n] = v;
                }
            }
        }
    }
}

// ---------------------------------------------------------------------------
// Scores + safe softmax, fused. One wave per (b, h, 32-row q tile): each
// key-tile B fragment feeds 4 WMMAs (2 q-subtiles x 2 K-steps).
// Pass 1: WMMA over all 128 key tiles, per-row max in regs + 16-lane shfl_xor.
// Pass 2: recompute, e = exp(s - max), row sums, store UNNORMALIZED bf16 attn
// to [b][q][h][k]; store 1/rowsum to invsum[b][h][q].
// ---------------------------------------------------------------------------
__global__ __launch_bounds__(256) void k_scores_softmax(const __bf16* __restrict__ Qp,
                                                        const __bf16* __restrict__ Kp,
                                                        __bf16* __restrict__ attn,
                                                        float* __restrict__ invsum) {
    int w    = (blockIdx.x * blockDim.x + threadIdx.x) >> 5;    // 0..2047
    int lane = threadIdx.x & 31;
    int lcol = lane & 15;
    int lhi  = lane >> 4;
    int qt = w & 63;            // 64 groups of 32 q rows
    int h  = (w >> 6) & 15;
    int b  = w >> 10;
    int q0 = qt * 32;

    // Q fragments: 2 q-subtiles x 2 K-steps (K-dim 64 -> 2 wmma steps)
    bf16x16 a[2][2];
    #pragma unroll
    for (int mi = 0; mi < 2; ++mi) {
        const __bf16* qrow = Qp + (size_t)(b * SEQ + q0 + mi * 16 + lcol) * EMBED + h * HDIM;
        a[mi][0] = load_frag(qrow + lhi * 16,      qrow + lhi * 16 + 8);
        a[mi][1] = load_frag(qrow + 32 + lhi * 16, qrow + 32 + lhi * 16 + 8);
    }

    const float scale = 0.125f;  // 1/sqrt(64)

    float rmax[2][8];
    #pragma unroll
    for (int mi = 0; mi < 2; ++mi)
        #pragma unroll
        for (int r = 0; r < 8; ++r) rmax[mi][r] = -1e30f;

    // -------- pass 1: row max --------
    for (int t = 0; t < SEQ / 16; ++t) {
        const __bf16* krow = Kp + (size_t)(b * SEQ + t * 16 + lcol) * EMBED + h * HDIM + lhi * 16;
        bf16x16 b0 = load_frag(krow,      krow + 8);
        bf16x16 b1 = load_frag(krow + 32, krow + 40);
        #pragma unroll
        for (int mi = 0; mi < 2; ++mi) {
            f32x8 c = {};
            c = wmma_bf16(a[mi][0], b0, c);
            c = wmma_bf16(a[mi][1], b1, c);
            #pragma unroll
            for (int r = 0; r < 8; ++r) rmax[mi][r] = fmaxf(rmax[mi][r], c[r] * scale);
        }
    }
    #pragma unroll
    for (int mask = 1; mask < 16; mask <<= 1)
        #pragma unroll
        for (int mi = 0; mi < 2; ++mi)
            #pragma unroll
            for (int r = 0; r < 8; ++r)
                rmax[mi][r] = fmaxf(rmax[mi][r], __shfl_xor(rmax[mi][r], mask, 32));

    // -------- pass 2: exp, sums, store --------
    float rsum[2][8];
    #pragma unroll
    for (int mi = 0; mi < 2; ++mi)
        #pragma unroll
        for (int r = 0; r < 8; ++r) rsum[mi][r] = 0.0f;

    for (int t = 0; t < SEQ / 16; ++t) {
        const __bf16* krow = Kp + (size_t)(b * SEQ + t * 16 + lcol) * EMBED + h * HDIM + lhi * 16;
        bf16x16 b0 = load_frag(krow,      krow + 8);
        bf16x16 b1 = load_frag(krow + 32, krow + 40);
        #pragma unroll
        for (int mi = 0; mi < 2; ++mi) {
            f32x8 c = {};
            c = wmma_bf16(a[mi][0], b0, c);
            c = wmma_bf16(a[mi][1], b1, c);
            size_t abase = ((size_t)(b * SEQ + q0 + mi * 16) * HEADS + h) * SEQ;
            #pragma unroll
            for (int r = 0; r < 8; ++r) {
                float e = __expf(c[r] * scale - rmax[mi][r]);
                rsum[mi][r] += e;
                int m = r + lhi * 8;                          // C-tile row
                attn[abase + (size_t)m * HEADS * SEQ + t * 16 + lcol] = (__bf16)e;
            }
        }
    }
    #pragma unroll
    for (int mask = 1; mask < 16; mask <<= 1)
        #pragma unroll
        for (int mi = 0; mi < 2; ++mi)
            #pragma unroll
            for (int r = 0; r < 8; ++r)
                rsum[mi][r] += __shfl_xor(rsum[mi][r], mask, 32);

    if (lcol == 0) {
        #pragma unroll
        for (int mi = 0; mi < 2; ++mi)
            #pragma unroll
            for (int r = 0; r < 8; ++r)
                invsum[(size_t)(b * HEADS + h) * SEQ + q0 + mi * 16 + r + lhi * 8] =
                    1.0f / rsum[mi][r];
    }
}

// ---------------------------------------------------------------------------
// Head mixing, IN PLACE on attn[b][q][h][k]. Applies 1/rowsum (normalization)
// then mixed[g] = sum_h softmax(head_mixing)[g][h] * attn_norm[h].
// One 256-thread block per (b,q); each thread owns unique k -> in-place safe.
// ---------------------------------------------------------------------------
__global__ __launch_bounds__(256) void k_mix(const float* __restrict__ head_mixing,
                                             const float* __restrict__ invsum,
                                             __bf16* __restrict__ attn) {
    __shared__ float smix[16][16];
    __shared__ float sinv[16];
    int bq = blockIdx.x;
    int b = bq >> 11, q = bq & 2047;
    int tid = threadIdx.x;

    if (tid < 16) {
        float row[16], mx = -1e30f;
        #pragma unroll
        for (int h = 0; h < 16; ++h) { row[h] = head_mixing[tid * 16 + h]; mx = fmaxf(mx, row[h]); }
        float s = 0.0f;
        #pragma unroll
        for (int h = 0; h < 16; ++h) { row[h] = __expf(row[h] - mx); s += row[h]; }
        float inv = 1.0f / s;
        #pragma unroll
        for (int h = 0; h < 16; ++h) smix[tid][h] = row[h] * inv;
        sinv[tid] = invsum[(size_t)(b * HEADS + tid) * SEQ + q];
    }
    __syncthreads();

    size_t base = (size_t)(b * SEQ + q) * HEADS * SEQ;
    for (int k = tid; k < SEQ; k += 256) {
        float a[16];
        #pragma unroll
        for (int h = 0; h < 16; ++h)
            a[h] = (float)attn[base + (size_t)h * SEQ + k] * sinv[h];
        float m[16];
        #pragma unroll
        for (int g = 0; g < 16; ++g) {
            float acc = 0.0f;
            #pragma unroll
            for (int h = 0; h < 16; ++h) acc += smix[g][h] * a[h];
            m[g] = acc;
        }
        #pragma unroll
        for (int g = 0; g < 16; ++g)
            attn[base + (size_t)g * SEQ + k] = (__bf16)m[g];
    }
}

// ---------------------------------------------------------------------------
// ctx = mixedAttn(Sq x Sk) @ V(Sk x 64) per (b,g). B operand = V^T [d][s].
// Register-blocked: each wave computes 32 q-rows x full 64-wide head dim
// (2 m-subtiles x 4 n-subtiles = 8 WMMAs per 32-wide K step).
// attn streams from HBM exactly once -> prefetch ahead (global_prefetch_b8).
// Writes ctx bf16 [b][q][g*64+d] (row-major activation for the O projection).
// ---------------------------------------------------------------------------
__global__ __launch_bounds__(256) void k_ctx_gemm(const __bf16* __restrict__ attn,
                                                  const __bf16* __restrict__ Vt,
                                                  __bf16* __restrict__ ctx) {
    int w    = (blockIdx.x * blockDim.x + threadIdx.x) >> 5;    // 0..2047
    int lane = threadIdx.x & 31;
    int lcol = lane & 15;
    int lhi  = lane >> 4;
    int qt = w & 63;            // 64 groups of 32 q rows
    int g  = (w >> 6) & 15;
    int b  = w >> 10;

    const __bf16* arow[2];
    #pragma unroll
    for (int mi = 0; mi < 2; ++mi)
        arow[mi] = attn + ((size_t)((b * SEQ + qt * 32 + mi * 16 + lcol) * HEADS + g)) * SEQ;
    const __bf16* brow[4];
    #pragma unroll
    for (int j = 0; j < 4; ++j)
        brow[j] = Vt + ((size_t)((b * HEADS + g) * HDIM + j * 16 + lcol)) * SEQ;

    f32x8 c[2][4] = {};
    for (int kk = 0; kk < SEQ; kk += 32) {
        // prefetch the streaming attn operand a few K-steps ahead
        if (kk + 256 < SEQ) {
            __builtin_prefetch(arow[0] + kk + 256, 0, 0);
            __builtin_prefetch(arow[1] + kk + 256, 0, 0);
        }
        bf16x16 a0 = load_frag(arow[0] + kk + lhi * 16, arow[0] + kk + lhi * 16 + 8);
        bf16x16 a1 = load_frag(arow[1] + kk + lhi * 16, arow[1] + kk + lhi * 16 + 8);
        #pragma unroll
        for (int j = 0; j < 4; ++j) {
            bf16x16 bf = load_frag(brow[j] + kk + lhi * 16, brow[j] + kk + lhi * 16 + 8);
            c[0][j] = wmma_bf16(a0, bf, c[0][j]);
            c[1][j] = wmma_bf16(a1, bf, c[1][j]);
        }
    }

    #pragma unroll
    for (int mi = 0; mi < 2; ++mi) {
        #pragma unroll
        for (int j = 0; j < 4; ++j) {
            int d = j * 16 + lcol;
            #pragma unroll
            for (int r = 0; r < 8; ++r) {
                int q = qt * 32 + mi * 16 + r + lhi * 8;
                ctx[(size_t)(b * SEQ + q) * EMBED + g * HDIM + d] = (__bf16)c[mi][j][r];
            }
        }
    }
}

// ---------------------------------------------------------------------------
// Host launch
// ---------------------------------------------------------------------------
extern "C" void kernel_launch(void* const* d_in, const int* in_sizes, int n_in,
                              void* d_out, int out_size, void* d_ws, size_t ws_size,
                              hipStream_t stream) {
    const float* query = (const float*)d_in[0];
    const float* key_  = (const float*)d_in[1];
    const float* value = (const float*)d_in[2];
    const float* Wq    = (const float*)d_in[3];
    const float* bq    = (const float*)d_in[4];
    const float* Wk    = (const float*)d_in[5];
    const float* bk    = (const float*)d_in[6];
    const float* Wv    = (const float*)d_in[7];
    const float* bv    = (const float*)d_in[8];
    const float* hmix  = (const float*)d_in[9];
    const float* Wo    = (const float*)d_in[10];
    const float* bo    = (const float*)d_in[11];

    char* p = (char*)d_ws;
    auto alloc = [&](size_t bytes) -> char* {
        char* r = p;
        p += (bytes + 255) & ~(size_t)255;
        return r;
    };

    const size_t ACT   = (size_t)MROWS * EMBED;                 // 4M elements
    const size_t WSZ   = (size_t)EMBED * EMBED;                 // 1M elements
    const size_t ATTNE = (size_t)BATCH * SEQ * HEADS * SEQ;     // 134M elements

    __bf16* Xq  = (__bf16*)alloc(ACT * 2);
    __bf16* Xk  = (__bf16*)alloc(ACT * 2);
    __bf16* Xv  = (__bf16*)alloc(ACT * 2);
    __bf16* WtQ = (__bf16*)alloc(WSZ * 2);
    __bf16* WtK = (__bf16*)alloc(WSZ * 2);
    __bf16* WtV = (__bf16*)alloc(WSZ * 2);
    __bf16* WtO = (__bf16*)alloc(WSZ * 2);
    __bf16* Qp  = (__bf16*)alloc(ACT * 2);
    __bf16* Kp  = (__bf16*)alloc(ACT * 2);
    __bf16* Vt  = (__bf16*)alloc(ACT * 2);
    __bf16* ctx = (__bf16*)alloc(ACT * 2);
    float*  inv = (float*) alloc((size_t)BATCH * HEADS * SEQ * 4);
    __bf16* att = (__bf16*)alloc(ATTNE * 2);                    // 256 MB

    // 1) activations fp32 -> bf16
    k_f32_to_bf16<<<4096, 256, 0, stream>>>(query, Xq, (int)ACT);
    k_f32_to_bf16<<<4096, 256, 0, stream>>>(key_,  Xk, (int)ACT);
    k_f32_to_bf16<<<4096, 256, 0, stream>>>(value, Xv, (int)ACT);

    // 2) weights -> transposed bf16 [N][K]
    dim3 tg(32, 32), tb(32, 8);
    k_transpose_bf16<<<tg, tb, 0, stream>>>(Wq, WtQ);
    k_transpose_bf16<<<tg, tb, 0, stream>>>(Wk, WtK);
    k_transpose_bf16<<<tg, tb, 0, stream>>>(Wv, WtV);
    k_transpose_bf16<<<tg, tb, 0, stream>>>(Wo, WtO);

    // 3) Q/K/V projections (WMMA, 32x64 per wave). 2048 waves -> 256 blocks.
    k_proj_gemm<0><<<256, 256, 0, stream>>>(Xq, WtQ, bq, (void*)Qp);
    k_proj_gemm<0><<<256, 256, 0, stream>>>(Xk, WtK, bk, (void*)Kp);
    k_proj_gemm<1><<<256, 256, 0, stream>>>(Xv, WtV, bv, (void*)Vt);   // writes V^T

    // 4) scores + softmax (WMMA, fused, register-resident). 2048 waves.
    k_scores_softmax<<<256, 256, 0, stream>>>(Qp, Kp, att, inv);

    // 5) head mixing (+ normalization), in place.
    k_mix<<<BATCH * SEQ, 256, 0, stream>>>(hmix, inv, att);

    // 6) context GEMM (WMMA, 32x64 per wave). 2048 waves.
    k_ctx_gemm<<<256, 256, 0, stream>>>(att, Vt, ctx);

    // 7) output projection -> fp32 d_out (WMMA).
    k_proj_gemm<2><<<256, 256, 0, stream>>>(ctx, WtO, bo, d_out);
}
